// WindowAttention_8701603742149
// MI455X (gfx1250) — compile-verified, matrix operand order checked by reading, not run on hardware
//
#include <hip/hip_runtime.h>
#include <hip/hip_bf16.h>
#include <stdint.h>

// ---------------------------------------------------------------------------
// SwinV2 3D window attention for MI455X (gfx1250), wave32 + WMMA f16 16x16x32.
// Pipeline: prep (f16 weights, bias gather, zero-pad) -> QKV GEMM + cosine
// normalize -> flash attention (online softmax, 1 WMMA per S tile since
// head_dim == 32 == WMMA K) -> output projection GEMM.
// Intermediates kept in f16 so QKV (101MB) + attn-out (34MB) live in the
// 192MB L2; HBM traffic ~ x + out only.
// ---------------------------------------------------------------------------

#define BW_   512
#define NTOK  343
#define NP    352          // padded rows (22 tiles of 16)
#define DIM   96
#define HEADS 3
#define HD    32
#define NWIN  64
#define MROWS (BW_ * NTOK)     // 175616, divisible by 16
#define MTILES (MROWS / 16)    // 10976

typedef _Float16 f16;
typedef __attribute__((ext_vector_type(16))) _Float16 v16h;
typedef __attribute__((ext_vector_type(8)))  float    v8f;

union Frag {
    v16h      v;
    uint32_t  u[8];
    _Float16  e[16];
};

// K-offset of element pair p (halves 2p,2p+1) for lane-group g, per the CDNA5
// 16-bit 16x32 operand layout: g=0 -> {0,2,4,6,16,18,20,22}, g=1 -> +8.
__device__ __forceinline__ int kpat(int p, int g) {
    return ((p & 3) << 1) + ((p >> 2) << 4) + (g << 3);
}

__device__ __forceinline__ v8f wmma16(const Frag& a, const Frag& b, v8f c) {
    return __builtin_amdgcn_wmma_f32_16x16x32_f16(false, a.v, false, b.v,
                                                  (short)0, c, false, false);
}

// butterfly reductions across the 16-lane half-group (xor offsets stay inside)
__device__ __forceinline__ float grp_max(float v) {
    v = fmaxf(v, __shfl_xor(v, 1, 32));
    v = fmaxf(v, __shfl_xor(v, 2, 32));
    v = fmaxf(v, __shfl_xor(v, 4, 32));
    v = fmaxf(v, __shfl_xor(v, 8, 32));
    return v;
}
__device__ __forceinline__ float grp_sum(float v) {
    v += __shfl_xor(v, 1, 32);
    v += __shfl_xor(v, 2, 32);
    v += __shfl_xor(v, 4, 32);
    v += __shfl_xor(v, 8, 32);
    return v;
}

// ---------------------------------------------------------------------------
// Prep kernels
// ---------------------------------------------------------------------------
__global__ void k_cvt_w(const float* __restrict__ wq, const float* __restrict__ wp,
                        f16* __restrict__ wq16, f16* __restrict__ wp16) {
    int i = blockIdx.x * 256 + threadIdx.x;
    if (i < 288 * DIM) wq16[i] = (f16)wq[i];
    if (i < DIM * DIM) wp16[i] = (f16)wp[i];
}

__global__ void k_bias(const float* __restrict__ table, const int* __restrict__ rpi,
                       float* __restrict__ bias) {
    int ij = blockIdx.x * 256 + threadIdx.x;
    if (ij < NTOK * NTOK) {
        int idx = rpi[ij];
        #pragma unroll
        for (int h = 0; h < HEADS; ++h)
            bias[h * NTOK * NTOK + ij] = table[idx * HEADS + h];
    }
}

__global__ void k_zero(uint32_t* __restrict__ p, long long n) {
    long long i = (long long)blockIdx.x * blockDim.x + threadIdx.x;
    long long stride = (long long)gridDim.x * blockDim.x;
    for (; i < n; i += stride) p[i] = 0u;
}

// ---------------------------------------------------------------------------
// Phase 1: qkv = x @ Wqkv^T + b ; L2-normalize q,k rows; pack f16 operands.
// One wave per 16-row tile. Q,K: [b][h][tok(352)][32] row-major.
// V stored transposed: [b][h][d(32)][tok(352)].
// ---------------------------------------------------------------------------
__global__ __launch_bounds__(32) void k_qkv(
    const float* __restrict__ x, const f16* __restrict__ w,
    const float* __restrict__ bq,
    f16* __restrict__ Q, f16* __restrict__ K, f16* __restrict__ Vt) {
    const int L  = threadIdx.x;
    const int lm = L & 15, g = L >> 4;
    const int mt = blockIdx.x;

    // A fragments: rows mt*16+lm of x, converted f32 -> f16
    Frag a[3];
    const float* xr = x + (size_t)(mt * 16 + lm) * DIM;
    #pragma unroll
    for (int kk = 0; kk < 3; ++kk)
        #pragma unroll
        for (int p = 0; p < 8; ++p) {
            int kq = kk * 32 + kpat(p, g);
            a[kk].e[2 * p]     = (f16)xr[kq];
            a[kk].e[2 * p + 1] = (f16)xr[kq + 1];
        }

    #pragma unroll
    for (int c = 0; c < 9; ++c) {           // 32-col chunk: s = c/3, h = c%3
        v8f acc0 = {}, acc1 = {};
        #pragma unroll
        for (int kk = 0; kk < 3; ++kk) {
            Frag b0, b1;
            const int n0 = c * 32 + lm, n1 = n0 + 16;
            #pragma unroll
            for (int p = 0; p < 8; ++p) {
                int kq = kk * 32 + kpat(p, g);
                b0.u[p] = *(const uint32_t*)(w + n0 * DIM + kq);
                b1.u[p] = *(const uint32_t*)(w + n1 * DIM + kq);
            }
            acc0 = wmma16(a[kk], b0, acc0);
            acc1 = wmma16(a[kk], b1, acc1);
        }
        const float bb0 = bq[c * 32 + lm];
        const float bb1 = bq[c * 32 + 16 + lm];
        #pragma unroll
        for (int r = 0; r < 8; ++r) { acc0[r] += bb0; acc1[r] += bb1; }

        const int s = c / 3, h = c % 3;
        if (s < 2) {                         // cosine-attention row normalize
            #pragma unroll
            for (int r = 0; r < 8; ++r) {
                float ss = grp_sum(acc0[r] * acc0[r] + acc1[r] * acc1[r]);
                float inv = 1.0f / fmaxf(sqrtf(ss), 1e-12f);
                acc0[r] *= inv; acc1[r] *= inv;
            }
        }
        #pragma unroll
        for (int r = 0; r < 8; ++r) {
            int fr  = mt * 16 + r + 8 * g;   // C layout: M = r + 8*(lane/16)
            int b   = fr / NTOK;
            int tok = fr - b * NTOK;
            size_t bh = (size_t)(b * HEADS + h);
            if (s == 0) {
                f16* d = Q + bh * NP * HD + (size_t)tok * HD;
                d[lm] = (f16)acc0[r]; d[16 + lm] = (f16)acc1[r];
            } else if (s == 1) {
                f16* d = K + bh * NP * HD + (size_t)tok * HD;
                d[lm] = (f16)acc0[r]; d[16 + lm] = (f16)acc1[r];
            } else {
                f16* d = Vt + bh * HD * NP;
                d[(size_t)lm * NP + tok]        = (f16)acc0[r];
                d[(size_t)(16 + lm) * NP + tok] = (f16)acc1[r];
            }
        }
    }
}

// ---------------------------------------------------------------------------
// Phase 2: flash attention per (window,head). 8 waves; each wave owns 16
// q-rows, streams K/V in chunks of 32 tokens with online softmax.
// ---------------------------------------------------------------------------
__global__ __launch_bounds__(256) void k_attn(
    const f16* __restrict__ Q, const f16* __restrict__ K,
    const f16* __restrict__ Vt, const float* __restrict__ bias,
    const float* __restrict__ mask, const float* __restrict__ logit_scale,
    f16* __restrict__ AO) {
    __shared__ _Float16 pscr[8][16][34];     // per-wave P transpose scratch

    const int b  = blockIdx.x / HEADS;
    const int h  = blockIdx.x % HEADS;
    const int wv = threadIdx.x >> 5;
    const int L  = threadIdx.x & 31;
    const int lm = L & 15, g = L >> 4;

    const float scale = __expf(fminf(logit_scale[h], 4.6051702f /* ln 100 */));
    const f16* Qb = Q  + (size_t)(b * HEADS + h) * NP * HD;
    const f16* Kb = K  + (size_t)(b * HEADS + h) * NP * HD;
    const f16* Vb = Vt + (size_t)(b * HEADS + h) * HD * NP;
    const float* biash = bias + (size_t)h * NTOK * NTOK;
    const float* maskw = mask + (size_t)(b % NWIN) * NTOK * NTOK;

    for (int rb = wv; rb < 22; rb += 8) {
        Frag aq;
        {
            const int row = rb * 16 + lm;    // padded rows are zero
            #pragma unroll
            for (int p = 0; p < 8; ++p)
                aq.u[p] = *(const uint32_t*)(Qb + (size_t)row * HD + kpat(p, g));
        }
        v8f o0 = {}, o1 = {};
        float m8[8], s8[8];
        #pragma unroll
        for (int r = 0; r < 8; ++r) { m8[r] = -1e38f; s8[r] = 0.0f; }

        for (int j = 0; j < 11; ++j) {       // 32-token chunks of K/V
            const int m0 = j * 32;
            Frag bk0, bk1;
            {
                const int t0 = m0 + lm, t1 = t0 + 16;
                #pragma unroll
                for (int p = 0; p < 8; ++p) {
                    int kq = kpat(p, g);
                    bk0.u[p] = *(const uint32_t*)(Kb + (size_t)t0 * HD + kq);
                    bk1.u[p] = *(const uint32_t*)(Kb + (size_t)t1 * HD + kq);
                }
            }
            v8f z = {};
            v8f s0 = wmma16(aq, bk0, z);     // S tile: rows=q, cols=k tokens
            v8f s1 = wmma16(aq, bk1, z);

            const int c0 = m0 + lm, c1 = c0 + 16;  // c0 always < 343
            const bool v1 = (c1 < NTOK);
            #pragma unroll
            for (int r = 0; r < 8; ++r) {
                int Mr = rb * 16 + r + 8 * g;
                int Mc = Mr < NTOK ? Mr : (NTOK - 1);
                const float* brow = biash + (size_t)Mc * NTOK;
                const float* mrow = maskw + (size_t)Mc * NTOK;
                s0[r] = s0[r] * scale + brow[c0] + mrow[c0];
                float b1v = v1 ? brow[c1] : 0.0f;
                float m1v = v1 ? mrow[c1] : 0.0f;
                s1[r] = v1 ? (s1[r] * scale + b1v + m1v) : -1e30f;
            }
            // online softmax update + stage P (f16) for layout transpose
            #pragma unroll
            for (int r = 0; r < 8; ++r) {
                float cm   = grp_max(fmaxf(s0[r], s1[r]));
                float nm   = fmaxf(m8[r], cm);
                float corr = __expf(m8[r] - nm);
                float p0   = __expf(s0[r] - nm);
                float p1   = __expf(s1[r] - nm);
                s8[r] = s8[r] * corr + grp_sum(p0 + p1);
                m8[r] = nm;
                o0[r] *= corr; o1[r] *= corr;
                pscr[wv][r + 8 * g][lm]      = (f16)p0;
                pscr[wv][r + 8 * g][16 + lm] = (f16)p1;
            }
            asm volatile("s_wait_dscnt 0" ::: "memory");  // cross-lane LDS vis
            Frag ap;                          // P as A-fragment [16 x 32]
            #pragma unroll
            for (int p = 0; p < 8; ++p)
                ap.u[p] = *(const uint32_t*)&pscr[wv][lm][kpat(p, g)];
            asm volatile("" ::: "memory");    // keep loads before next stores

            Frag bv0, bv1;                    // V^T tiles: B[k=m][n=d]
            #pragma unroll
            for (int p = 0; p < 8; ++p) {
                int kq = m0 + kpat(p, g);
                bv0.u[p] = *(const uint32_t*)(Vb + (size_t)lm * NP + kq);
                bv1.u[p] = *(const uint32_t*)(Vb + (size_t)(16 + lm) * NP + kq);
            }
            o0 = wmma16(ap, bv0, o0);
            o1 = wmma16(ap, bv1, o1);
        }
        #pragma unroll
        for (int r = 0; r < 8; ++r) {
            float inv = 1.0f / s8[r];
            int Mr = rb * 16 + r + 8 * g;
            if (Mr < NTOK) {
                f16* d = AO + ((size_t)b * NTOK + Mr) * DIM + h * HD;
                d[lm]      = (f16)(o0[r] * inv);
                d[16 + lm] = (f16)(o1[r] * inv);
            }
        }
    }
}

// ---------------------------------------------------------------------------
// Phase 3: out = attn_out @ Wproj^T + bproj  (M=175616, N=96, K=96)
// ---------------------------------------------------------------------------
__global__ __launch_bounds__(32) void k_proj(
    const f16* __restrict__ AO, const f16* __restrict__ wp,
    const float* __restrict__ bp, float* __restrict__ out) {
    const int L = threadIdx.x, lm = L & 15, g = L >> 4;
    const int mt = blockIdx.x;

    Frag a[3];
    const f16* ar = AO + (size_t)(mt * 16 + lm) * DIM;
    #pragma unroll
    for (int kk = 0; kk < 3; ++kk)
        #pragma unroll
        for (int p = 0; p < 8; ++p)
            a[kk].u[p] = *(const uint32_t*)(ar + kk * 32 + kpat(p, g));

    #pragma unroll
    for (int nt = 0; nt < 6; ++nt) {
        v8f acc = {};
        #pragma unroll
        for (int kk = 0; kk < 3; ++kk) {
            Frag bf;
            const int n = nt * 16 + lm;
            #pragma unroll
            for (int p = 0; p < 8; ++p)
                bf.u[p] = *(const uint32_t*)(wp + n * DIM + kk * 32 + kpat(p, g));
            acc = wmma16(a[kk], bf, acc);
        }
        const float bb = bp[nt * 16 + lm];
        #pragma unroll
        for (int r = 0; r < 8; ++r) {
            int fr = mt * 16 + r + 8 * g;
            out[(size_t)fr * DIM + nt * 16 + lm] = acc[r] + bb;
        }
    }
}

// ---------------------------------------------------------------------------
extern "C" void kernel_launch(void* const* d_in, const int* in_sizes, int n_in,
                              void* d_out, int out_size, void* d_ws, size_t ws_size,
                              hipStream_t stream) {
    (void)in_sizes; (void)n_in; (void)out_size; (void)ws_size;
    const float* x           = (const float*)d_in[0];
    const float* mask        = (const float*)d_in[1];
    const float* Wqkv        = (const float*)d_in[2];
    const float* bqkv        = (const float*)d_in[3];
    const float* Wproj       = (const float*)d_in[4];
    const float* bproj       = (const float*)d_in[5];
    const float* logit_scale = (const float*)d_in[6];
    const float* bias_table  = (const float*)d_in[7];
    const int*   rpi         = (const int*)d_in[8];
    float* out = (float*)d_out;

    // workspace layout (~133 MiB total)
    char* ws = (char*)d_ws;
    size_t off = 0;
    auto take = [&](size_t bytes) {
        void* p = ws + off;
        off = (off + bytes + 255) & ~(size_t)255;
        return p;
    };
    f16*   wq16 = (f16*)  take((size_t)288 * DIM * sizeof(f16));
    f16*   wp16 = (f16*)  take((size_t)DIM * DIM * sizeof(f16));
    float* bias = (float*)take((size_t)HEADS * NTOK * NTOK * sizeof(float));
    const size_t qkv_bytes = (size_t)BW_ * HEADS * NP * HD * sizeof(f16); // 34.6MB
    f16*   Qb   = (f16*)  take(qkv_bytes);
    f16*   Kb   = (f16*)  take(qkv_bytes);
    f16*   Vt   = (f16*)  take(qkv_bytes);
    f16*   AO   = (f16*)  take((size_t)MROWS * DIM * sizeof(f16));        // 33.7MB

    // prep
    k_cvt_w<<<(288 * DIM + 255) / 256, 256, 0, stream>>>(Wqkv, Wproj, wq16, wp16);
    k_bias<<<(NTOK * NTOK + 255) / 256, 256, 0, stream>>>(bias_table, rpi, bias);
    const long long zdw = (long long)(3 * qkv_bytes) / 4;  // zero Q,K,Vt (pad rows)
    k_zero<<<2048, 256, 0, stream>>>((uint32_t*)Qb, zdw);

    // pipeline
    k_qkv <<<MTILES, 32, 0, stream>>>(x, wq16, bqkv, Qb, Kb, Vt);
    k_attn<<<BW_ * HEADS, 256, 0, stream>>>(Qb, Kb, Vt, bias, mask, logit_scale, AO);
    k_proj<<<MTILES, 32, 0, stream>>>(AO, wp16, bproj, out);
}
